// MoeLayer_64630667870330
// MI455X (gfx1250) — matrix operand order repaired and checked by reading, not compiled, not run on hardware
//
#include <hip/hip_runtime.h>
#include <hip/hip_bf16.h>

typedef __attribute__((ext_vector_type(16))) __bf16 v16bf;
typedef __attribute__((ext_vector_type(8)))  float  v8f;
typedef __attribute__((ext_vector_type(4)))  unsigned int v4u;

#define LL 2048
#define BB 4
#define NTOK (LL * BB)      // 8192 tokens
#define DD 1024
#define FF 4096
#define EE 8
#define KK 2

union V16 { v16bf bf; v4u u[2]; };

__device__ __forceinline__ unsigned short f2bf(float f) {
    union { float f; unsigned int u; } v; v.f = f;
    unsigned int u = v.u;
    unsigned int r = (u + 0x7FFFu + ((u >> 16) & 1u)) >> 16;
    return (unsigned short)r;
}

__device__ __forceinline__ v8f wmma_bf16(const V16& a, const V16& b, v8f c) {
    return __builtin_amdgcn_wmma_f32_16x16x32_bf16(false, a.bf, false, b.bf,
                                                   (short)0, c, false, false);
}

// CDNA5 async copy global->LDS. INST_OFFSET is added to BOTH the global and
// the LDS address, so identity slab copies need only one base address each.
#define ASYNC_CP(l, g, OFF) \
    asm volatile("global_load_async_to_lds_b128 %0, %1, off offset:" #OFF \
                 :: "v"(l), "v"(g) : "memory")

// copy 4 KB slab (per wave): 8 x 16B per lane, 512B apart
#define SLAB_COPY_4K(l, g) do { \
    ASYNC_CP(l, g, 0);    ASYNC_CP(l, g, 512);  ASYNC_CP(l, g, 1024); \
    ASYNC_CP(l, g, 1536); ASYNC_CP(l, g, 2048); ASYNC_CP(l, g, 2560); \
    ASYNC_CP(l, g, 3072); ASYNC_CP(l, g, 3584); } while (0)

// copy 2 KB row (per thread-slice): 8 x 16B, 256B apart
#define ROW_COPY_2K(l, g) do { \
    ASYNC_CP(l, g, 0);    ASYNC_CP(l, g, 256);  ASYNC_CP(l, g, 512); \
    ASYNC_CP(l, g, 768);  ASYNC_CP(l, g, 1024); ASYNC_CP(l, g, 1280); \
    ASYNC_CP(l, g, 1536); ASYNC_CP(l, g, 1792); } while (0)

#define WAIT_ASYNC_8() asm volatile("s_wait_asynccnt 0x8" ::: "memory")
#define WAIT_ASYNC_0() asm volatile("s_wait_asynccnt 0x0" ::: "memory")

__device__ __forceinline__ unsigned lds_off(const void* p) {
    return (unsigned)(unsigned long long)p;   // low 32 bits = LDS byte offset
}

// ---------------------------------------------------------------- zero init
__global__ __launch_bounds__(256)
void zero_kernel(float* out, int* cnt, float* swsum) {
    size_t total = (size_t)NTOK * DD + 1;
    size_t stride = (size_t)gridDim.x * blockDim.x;
    for (size_t i = (size_t)blockIdx.x * blockDim.x + threadIdx.x; i < total; i += stride)
        out[i] = 0.0f;
    if (blockIdx.x == 0 && threadIdx.x < EE) {
        cnt[threadIdx.x] = 0;
        swsum[threadIdx.x] = 0.0f;
    }
}

// ------------------------------------------------- gl_task = task@Wg_task+bg
__global__ __launch_bounds__(32)
void task_gate_kernel(const float* task, const float* Wg_task, const float* bg_task,
                      float* glt) {
    int e = threadIdx.x;
    if (e < EE) {
        float s = 0.0f;
        for (int d = 0; d < DD; ++d) s += task[d] * Wg_task[d * EE + e];
        glt[e] = s + bg_task[e];
    }
}

// ------------------------------------------------------- fp32 -> bf16 tokens
__global__ __launch_bounds__(256)
void cvt_x_kernel(const float* x, unsigned short* Xbf) {
    size_t total = (size_t)NTOK * DD;
    size_t stride = (size_t)gridDim.x * blockDim.x;
    for (size_t i = (size_t)blockIdx.x * blockDim.x + threadIdx.x; i < total; i += stride)
        Xbf[i] = f2bf(x[i]);
}

// -------------- pack W1 [E,D,F] fp32 -> WMMA-B lane layout bf16 (dword each)
// dword index = (((e*(F/16)+fb)*(D/32)+kb)*32 + lane)*8 + j
// element: k = kb*32 + (lane>>4)*16 + j*2 (+0/1), col = fb*16 + (lane&15)
__global__ __launch_bounds__(256)
void pack_w1_kernel(const float* W1, unsigned int* W1p) {
    size_t total = (size_t)EE * DD * FF / 2;
    size_t stride = (size_t)gridDim.x * blockDim.x;
    for (size_t idx = (size_t)blockIdx.x * blockDim.x + threadIdx.x; idx < total; idx += stride) {
        int j    = (int)(idx & 7);
        int lane = (int)((idx >> 3) & 31);
        size_t t = idx >> 8;
        int kb = (int)(t % (DD / 32)); t /= (DD / 32);
        int fb = (int)(t % (FF / 16)); t /= (FF / 16);
        int e  = (int)t;
        int k   = kb * 32 + (lane >> 4) * 16 + j * 2;
        int col = fb * 16 + (lane & 15);
        float f0 = W1[((size_t)e * DD + k) * FF + col];
        float f1 = W1[((size_t)e * DD + k + 1) * FF + col];
        W1p[idx] = (unsigned int)f2bf(f0) | ((unsigned int)f2bf(f1) << 16);
    }
}

// -------------- pack W2 [E,F,D] fp32 -> WMMA-B lane layout bf16
// dword index = (((e*(D/16)+db)*(F/32)+kb2)*32 + lane)*8 + j
__global__ __launch_bounds__(256)
void pack_w2_kernel(const float* W2, unsigned int* W2p) {
    size_t total = (size_t)EE * FF * DD / 2;
    size_t stride = (size_t)gridDim.x * blockDim.x;
    for (size_t idx = (size_t)blockIdx.x * blockDim.x + threadIdx.x; idx < total; idx += stride) {
        int j    = (int)(idx & 7);
        int lane = (int)((idx >> 3) & 31);
        size_t t = idx >> 8;
        int kb2 = (int)(t % (FF / 32)); t /= (FF / 32);
        int db  = (int)(t % (DD / 16)); t /= (DD / 16);
        int e   = (int)t;
        int k   = kb2 * 32 + (lane >> 4) * 16 + j * 2;   // f index
        int col = db * 16 + (lane & 15);                 // d index
        float f0 = W2[((size_t)e * FF + k) * DD + col];
        float f1 = W2[((size_t)e * FF + k + 1) * DD + col];
        W2p[idx] = (unsigned int)f2bf(f0) | ((unsigned int)f2bf(f1) << 16);
    }
}

// ------------------------------------------------ gating + top-2 + aux stats
__global__ __launch_bounds__(64)
void gate_kernel(const float* x, const float* Wg_in, const float* bg_in,
                 const float* glt, const float* alpha_p,
                 int* toks, float* wts, int* cnt, float* swsum) {
    int t   = blockIdx.x;          // token
    int tid = threadIdx.x;
    __shared__ float red[64][EE];

    float a[EE];
#pragma unroll
    for (int e = 0; e < EE; ++e) a[e] = 0.0f;

    const float* xr = x + (size_t)t * DD;
    for (int d = tid; d < DD; d += 64) {
        float xv = xr[d];
        const float* wr = Wg_in + (size_t)d * EE;
#pragma unroll
        for (int e = 0; e < EE; ++e) a[e] += xv * wr[e];
    }
#pragma unroll
    for (int e = 0; e < EE; ++e) red[tid][e] = a[e];
    __syncthreads();
    for (int s = 32; s > 0; s >>= 1) {
        if (tid < s) {
#pragma unroll
            for (int e = 0; e < EE; ++e) red[tid][e] += red[tid + s][e];
        }
        __syncthreads();
    }
    if (tid == 0) {
        float alpha = *alpha_p;
        float lg[EE];
#pragma unroll
        for (int e = 0; e < EE; ++e) {
            float v = (1.0f - alpha) * (red[0][e] + bg_in[e]) + alpha * glt[e];
            if (!__builtin_isfinite(v)) v = 0.0f;
            lg[e] = v;
        }
        // full softmax for load-balance stats
        float mx = lg[0];
#pragma unroll
        for (int e = 1; e < EE; ++e) mx = fmaxf(mx, lg[e]);
        float se = 0.0f; float w[EE];
#pragma unroll
        for (int e = 0; e < EE; ++e) { w[e] = __expf(lg[e] - mx); se += w[e]; }
        float inv = 1.0f / se;
#pragma unroll
        for (int e = 0; e < EE; ++e) atomicAdd(&swsum[e], w[e] * inv);
        // top-2
        int e1 = 0;
#pragma unroll
        for (int e = 1; e < EE; ++e) if (lg[e] > lg[e1]) e1 = e;
        int e2 = (e1 == 0) ? 1 : 0;
#pragma unroll
        for (int e = 0; e < EE; ++e) if (e != e1 && lg[e] > lg[e2]) e2 = e;
        float z  = __expf(lg[e2] - lg[e1]);
        float w1 = 1.0f / (1.0f + z);
        float w2 = z * w1;
        int p1 = atomicAdd(&cnt[e1], 1);
        toks[e1 * NTOK + p1] = t; wts[e1 * NTOK + p1] = w1;
        int p2 = atomicAdd(&cnt[e2], 1);
        toks[e2 * NTOK + p2] = t; wts[e2 * NTOK + p2] = w2;
    }
}

// ----------------------------------------------------- fused expert FFN tile
// block = 256 threads (8 wave32), 16 tokens per block, full D output.
// Weights streamed via async global->LDS copies, double-buffered per wave.
__global__ __launch_bounds__(256)
void ffn_kernel(const unsigned short* Xbf,
                const unsigned short* W1p, const unsigned short* W2p,
                const float* b1, const float* b2,
                const int* toks, const float* wts, const int* cnt,
                float* out) {
    int e    = blockIdx.y;
    int tile = blockIdx.x;
    int n    = cnt[e];
    int base = tile * 16;
    if (base >= n) return;

    __shared__ unsigned short Xs[16 * DD];        // 32 KB token tile, bf16
    __shared__ unsigned short Hs[16 * 128];       // 4 KB  hidden chunk, bf16
    __shared__ unsigned short Wst[8 * 2 * 2048];  // 64 KB: per-wave 2 x 4KB buffers
    __shared__ int   tokS[16];
    __shared__ float wS[16];

    int tid  = threadIdx.x;
    int wave = tid >> 5;
    int lane = tid & 31;
    int lm   = lane & 15;
    int lg   = lane >> 4;
    unsigned lane16 = lane * 16;                  // byte offset within a slab

    if (tid < 16) {
        int gi = base + tid;
        int tk = 0; float wv = 0.0f;
        if (gi < n) { tk = toks[e * NTOK + gi]; wv = wts[e * NTOK + gi]; }
        tokS[tid] = tk; wS[tid] = wv;
    }
    __syncthreads();

    // ---- async-gather token rows into LDS (each thread: one 2KB-row slice)
    {
        int r   = tid >> 4;                       // token row 0..15
        int seg = tid & 15;                       // 16B segment within row
        unsigned l = lds_off(Xs) + (unsigned)r * 2048 + (unsigned)seg * 16;
        const char* g = (const char*)(Xbf + (size_t)tokS[r] * DD) + seg * 16;
        ROW_COPY_2K(l, g);
    }
    WAIT_ASYNC_0();
    __syncthreads();

    unsigned wst_base = lds_off(Wst) + wave * 8192;   // this wave's 8 KB region
    const unsigned short* wstS = Wst + wave * 4096;   // same region, ushort units

    v8f acc[8];
    v8f vz = {0.f, 0.f, 0.f, 0.f, 0.f, 0.f, 0.f, 0.f};
#pragma unroll
    for (int c = 0; c < 8; ++c) acc[c] = vz;

    for (int fc = 0; fc < FF / 128; ++fc) {
        // ============ GEMM1: H[16, wave's 16 cols] = X @ W1 chunk (K = D)
        int fb = fc * 8 + wave;                       // F/16 block
        const char* gW1 = (const char*)W1p +
            (((size_t)e * (FF / 16) + fb) * (DD / 32)) * 1024 + lane16;

        { unsigned l = wst_base + lane16; SLAB_COPY_4K(l, gW1); }   // slab 0 -> buf 0

        v8f h = vz;
        for (int sb = 0; sb < 8; ++sb) {              // 8 slabs x 4 kb-steps
            if (sb < 7) {                             // stage next slab
                unsigned l = wst_base + ((sb + 1) & 1) * 4096 + lane16;
                const char* g = gW1 + (size_t)(sb + 1) * 4096;
                SLAB_COPY_4K(l, g);
                WAIT_ASYNC_8();                       // slab sb resident
            } else {
                WAIT_ASYNC_0();
            }
            int buf = sb & 1;
#pragma unroll
            for (int q = 0; q < 4; ++q) {
                int kb = sb * 4 + q;
                V16 a, b;
                a.u[0] = *(const v4u*)(Xs + lm * DD + kb * 32 + lg * 16);
                a.u[1] = *(const v4u*)(Xs + lm * DD + kb * 32 + 8 + lg * 16);
                const unsigned short* wp = wstS + buf * 2048 + q * 512 + lane * 16;
                b.u[0] = *(const v4u*)wp;
                b.u[1] = *(const v4u*)(wp + 8);
                h = wmma_bf16(a, b, h);
            }
        }

        // overlap: stage GEMM2 slab for c=0 while doing SiLU + barrier
        const char* gW2base = (const char*)W2p +
            (((size_t)e * (DD / 16) + wave * 8) * (FF / 32) + (size_t)fc * 4) * 1024 + lane16;
        { unsigned l = wst_base + lane16; SLAB_COPY_4K(l, gW2base); }

        // bias + SiLU, write bf16 H chunk to LDS
        float bias = b1[e * FF + fc * 128 + wave * 16 + lm];
#pragma unroll
        for (int r = 0; r < 8; ++r) {
            float xv = h[r] + bias;
            float sv = xv / (1.0f + __expf(-xv));
            int m = r + lg * 8;
            Hs[m * 128 + wave * 16 + lm] = f2bf(sv);
        }
        __syncthreads();

        // ============ GEMM2: Y[16, wave's 128 cols] += H @ W2 chunk (K = 128)
        // A-fragments are shared by all 8 C-tiles: hoist into registers once.
        V16 a2[4];
#pragma unroll
        for (int kq = 0; kq < 4; ++kq) {
            a2[kq].u[0] = *(const v4u*)(Hs + lm * 128 + kq * 32 + lg * 16);
            a2[kq].u[1] = *(const v4u*)(Hs + lm * 128 + kq * 32 + 8 + lg * 16);
        }
#pragma unroll
        for (int c = 0; c < 8; ++c) {
            if (c < 7) {                              // stage slab for c+1
                unsigned l = wst_base + ((c + 1) & 1) * 4096 + lane16;
                const char* g = gW2base + (size_t)(c + 1) * (FF / 32) * 1024;
                SLAB_COPY_4K(l, g);
                WAIT_ASYNC_8();
            } else {
                WAIT_ASYNC_0();
            }
            int buf = c & 1;
#pragma unroll
            for (int kq = 0; kq < 4; ++kq) {
                V16 b;
                const unsigned short* wp = wstS + buf * 2048 + kq * 512 + lane * 16;
                b.u[0] = *(const v4u*)wp;
                b.u[1] = *(const v4u*)(wp + 8);
                acc[c] = wmma_bf16(a2[kq], b, acc[c]);
            }
        }
        __syncthreads();                              // Hs reuse next chunk
    }

    // ---- epilogue: out[token, d] += comb * (y + b2)
#pragma unroll
    for (int c = 0; c < 8; ++c) {
        int d0 = wave * 128 + c * 16 + lm;
        float b2v = b2[e * DD + d0];
#pragma unroll
        for (int r = 0; r < 8; ++r) {
            int m = r + lg * 8;
            if (base + m < n) {
                float val = (acc[c][r] + b2v) * wS[m];
                atomicAdd(out + (size_t)tokS[m] * DD + d0, val);
            }
        }
    }
}

// --------------------------------------------------------------- l_aux scalar
__global__ void aux_kernel(const int* cnt, const float* swsum, float* out) {
    if (threadIdx.x == 0 && blockIdx.x == 0) {
        float s = 0.0f;
        float invN = 1.0f / (float)NTOK;
        for (int e = 0; e < EE; ++e)
            s += (swsum[e] * invN) * ((float)cnt[e] * invN);
        out[(size_t)NTOK * DD] = s;   // * E * (1/E) cancels; LB_WEIGHT = 1
    }
}

extern "C" void kernel_launch(void* const* d_in, const int* in_sizes, int n_in,
                              void* d_out, int out_size, void* d_ws, size_t ws_size,
                              hipStream_t stream) {
    const float* inputs   = (const float*)d_in[0];
    const float* task     = (const float*)d_in[1];
    const float* Wg_in    = (const float*)d_in[2];
    const float* bg_in    = (const float*)d_in[3];
    const float* Wg_task  = (const float*)d_in[4];
    const float* bg_task  = (const float*)d_in[5];
    const float* alpha    = (const float*)d_in[6];
    const float* W1       = (const float*)d_in[7];
    const float* b1       = (const float*)d_in[8];
    const float* W2       = (const float*)d_in[9];
    const float* b2       = (const float*)d_in[10];
    float* out = (float*)d_out;

    char* ws = (char*)d_ws;
    size_t o = 0;
    unsigned short* Xbf = (unsigned short*)(ws + o); o += (size_t)NTOK * DD * 2;
    unsigned short* W1p = (unsigned short*)(ws + o); o += (size_t)EE * DD * FF * 2;
    unsigned short* W2p = (unsigned short*)(ws + o); o += (size_t)EE * FF * DD * 2;
    int*   toks  = (int*)  (ws + o); o += (size_t)EE * NTOK * 4;
    float* wts   = (float*)(ws + o); o += (size_t)EE * NTOK * 4;
    int*   cnt   = (int*)  (ws + o); o += 64;
    float* swsum = (float*)(ws + o); o += 64;
    float* glt   = (float*)(ws + o); o += 64;

    zero_kernel<<<2048, 256, 0, stream>>>(out, cnt, swsum);
    task_gate_kernel<<<1, 32, 0, stream>>>(task, Wg_task, bg_task, glt);
    cvt_x_kernel<<<4096, 256, 0, stream>>>(inputs, Xbf);
    pack_w1_kernel<<<8192, 256, 0, stream>>>(W1, (unsigned int*)W1p);
    pack_w2_kernel<<<8192, 256, 0, stream>>>(W2, (unsigned int*)W2p);
    gate_kernel<<<NTOK, 64, 0, stream>>>(inputs, Wg_in, bg_in, glt, alpha,
                                         toks, wts, cnt, swsum);
    ffn_kernel<<<dim3(NTOK / 16, EE), 256, 0, stream>>>(Xbf, W1p, W2p, b1, b2,
                                                        toks, wts, cnt, out);
    aux_kernel<<<1, 32, 0, stream>>>(cnt, swsum, out);
}
